// RGCN_17016660426944
// MI455X (gfx1250) — matrix operand order repaired and checked by reading, not compiled, not run on hardware
//
#include <hip/hip_runtime.h>

// ---------------------------------------------------------------------------
// RGCN 2-layer forward, MI455X (gfx1250, wave32, WMMA).
//
// Reassociation: segment_sum((edge_h + h[src]) @ W_rel) * norm + h @ W_loop
//              = (segment_sum(edge_h + h[src]) * norm) @ W_rel + h @ W_loop
// => scatter raw 128-float vectors with f32 atomics, then ONE fused GEMM per
//    layer ([agg*norm | h] @ [W_rel ; W_loop]) as two bf16 WMMA chains into a
//    shared f32 accumulator. Weights staged in LDS (64 KB) per workgroup.
// Bandwidth-bound (~0.8 GB/layer @ 23.3 TB/s); 26 GFLOP total is negligible.
// ---------------------------------------------------------------------------

#define RGCN_D 128
#define RRELU_SLOPE 0.2291666666666667f

typedef __attribute__((ext_vector_type(16))) __bf16 v16bf;
typedef __attribute__((ext_vector_type(8)))  float  v8f;

union AFrag { v16bf v; __bf16 b[16]; };
union BFrag { v16bf v; uint4  u[2]; };

__device__ __forceinline__ unsigned short f2bf(float f) {
  union { float f; unsigned u; } in; in.f = f;
  unsigned r = in.u + 0x7FFFu + ((in.u >> 16) & 1u);   // RNE (repack only)
  return (unsigned short)(r >> 16);
}

// ----------------------------- zero scratch --------------------------------
__global__ void rgcn_zero(float* __restrict__ p, long n) {
  long i = (long)blockIdx.x * blockDim.x + threadIdx.x;
  if (i < n) p[i] = 0.0f;
}

// ------------------- edge scatter: agg[dst] += rel[rid] + h[src] -----------
// 32 threads per edge, float4 per thread (128 floats / edge).
__global__ void rgcn_scatter(const float* __restrict__ h,
                             const float* __restrict__ rel,
                             const int* __restrict__ src,
                             const int* __restrict__ dst,
                             const int* __restrict__ rid,
                             float* __restrict__ agg, int E) {
  int t = blockIdx.x * blockDim.x + threadIdx.x;
  int e = t >> 5;
  if (e >= E) return;
  int c = (t & 31) << 2;
  int s = src[e], d = dst[e], r = rid[e];
  float4 hv = *reinterpret_cast<const float4*>(h   + (long)s * RGCN_D + c);
  float4 rv = *reinterpret_cast<const float4*>(rel + (long)r * RGCN_D + c);
  float* o = agg + (long)d * RGCN_D + c;
  atomicAdd(o + 0, hv.x + rv.x);
  atomicAdd(o + 1, hv.y + rv.y);
  atomicAdd(o + 2, hv.z + rv.z);
  atomicAdd(o + 3, hv.w + rv.w);
}

// --------------- repack W_rel/W_loop into WMMA B-fragment bf16 -------------
// packed[((w*4+k)*8+n)*32 + lane][16 halfs], contiguous 32B per lane.
// B (32x16) layout: col = lane%16; K = (lane/16)*8 + elem%8 + 16*(elem/8)  (per K-step k).
__global__ void rgcn_repack(const float* __restrict__ Wrel,
                            const float* __restrict__ Wloop,
                            unsigned short* __restrict__ packed) {
  int t = blockIdx.x * blockDim.x + threadIdx.x;
  if (t >= 2 * 4 * 8 * 32 * 16) return;
  int e    = t & 15;
  int lane = (t >> 4) & 31;
  int n    = (t >> 9) & 7;
  int k    = (t >> 12) & 3;
  int w    = (t >> 14) & 1;
  int col = n * 16 + (lane & 15);
  int kg  = k * 32 + (lane >> 4) * 8 + (e & 7) + (e >> 3) * 16;
  const float* W = w ? Wloop : Wrel;
  packed[t] = f2bf(W[kg * RGCN_D + col]);
}

// ---------------- fused GEMM: out = lrelu(agg*norm @ Wr + h @ Wl) ----------
// 8 waves / block; one 16-row tile per wave; weights staged once in LDS.
// 4 K-steps x 8 N-tiles x 2 weights = 64 v_wmma_f32_16x16x32_bf16 per wave.
__global__ void __launch_bounds__(256)
rgcn_gemm(const float* __restrict__ agg,
          const float* __restrict__ h,
          const float* __restrict__ norm,
          const unsigned short* __restrict__ packed,
          float* __restrict__ out, int ntiles) {
  __shared__ unsigned short wlds[2 * 4 * 8 * 32 * 16];   // 64 KB

  // Cooperative LDS fill: 4096 uint4 over 256 threads (b128 in, b128 out).
  {
    const uint4* gsrc = reinterpret_cast<const uint4*>(packed);
    uint4*       ldst = reinterpret_cast<uint4*>(wlds);
#pragma unroll
    for (int i = 0; i < 16; ++i)
      ldst[i * 256 + threadIdx.x] = gsrc[i * 256 + threadIdx.x];
  }
  __syncthreads();

  int warp = threadIdx.x >> 5;
  int lane = threadIdx.x & 31;
  int tile = blockIdx.x * 8 + warp;
  if (tile >= ntiles) return;            // wave-uniform exit; no barrier after
  int half = lane >> 4;
  int row  = tile * 16 + (lane & 15);
  float nrm = norm[row];
  const float* aRow = agg + (long)row * RGCN_D;
  const float* hRow = h   + (long)row * RGCN_D;

  v8f acc[8] = {};

#pragma unroll
  for (int k = 0; k < 4; ++k) {
    // A layout per lane: elems 0..7  -> K = k*32 + 8*half + 0..7  (contiguous)
    //                    elems 8..15 -> K = k*32 + 16 + 8*half + 0..7
    const float* aP = aRow + k * 32 + 8 * half;
    const float* hP = hRow + k * 32 + 8 * half;
    float4 ga0 = *reinterpret_cast<const float4*>(aP);
    float4 ga1 = *reinterpret_cast<const float4*>(aP + 4);
    float4 ga2 = *reinterpret_cast<const float4*>(aP + 16);
    float4 ga3 = *reinterpret_cast<const float4*>(aP + 20);
    float4 gh0 = *reinterpret_cast<const float4*>(hP);
    float4 gh1 = *reinterpret_cast<const float4*>(hP + 4);
    float4 gh2 = *reinterpret_cast<const float4*>(hP + 16);
    float4 gh3 = *reinterpret_cast<const float4*>(hP + 20);

    AFrag a1, a2;
    a1.b[0]  = (__bf16)(ga0.x * nrm); a1.b[1]  = (__bf16)(ga0.y * nrm);
    a1.b[2]  = (__bf16)(ga0.z * nrm); a1.b[3]  = (__bf16)(ga0.w * nrm);
    a1.b[4]  = (__bf16)(ga1.x * nrm); a1.b[5]  = (__bf16)(ga1.y * nrm);
    a1.b[6]  = (__bf16)(ga1.z * nrm); a1.b[7]  = (__bf16)(ga1.w * nrm);
    a1.b[8]  = (__bf16)(ga2.x * nrm); a1.b[9]  = (__bf16)(ga2.y * nrm);
    a1.b[10] = (__bf16)(ga2.z * nrm); a1.b[11] = (__bf16)(ga2.w * nrm);
    a1.b[12] = (__bf16)(ga3.x * nrm); a1.b[13] = (__bf16)(ga3.y * nrm);
    a1.b[14] = (__bf16)(ga3.z * nrm); a1.b[15] = (__bf16)(ga3.w * nrm);
    a2.b[0]  = (__bf16)gh0.x; a2.b[1]  = (__bf16)gh0.y;
    a2.b[2]  = (__bf16)gh0.z; a2.b[3]  = (__bf16)gh0.w;
    a2.b[4]  = (__bf16)gh1.x; a2.b[5]  = (__bf16)gh1.y;
    a2.b[6]  = (__bf16)gh1.z; a2.b[7]  = (__bf16)gh1.w;
    a2.b[8]  = (__bf16)gh2.x; a2.b[9]  = (__bf16)gh2.y;
    a2.b[10] = (__bf16)gh2.z; a2.b[11] = (__bf16)gh2.w;
    a2.b[12] = (__bf16)gh3.x; a2.b[13] = (__bf16)gh3.y;
    a2.b[14] = (__bf16)gh3.z; a2.b[15] = (__bf16)gh3.w;

#pragma unroll
    for (int n = 0; n < 8; ++n) {
      BFrag b1, b2;
      const uint4* l1 = reinterpret_cast<const uint4*>(
          wlds + (((0 * 4 + k) * 8 + n) * 32 + lane) * 16);
      const uint4* l2 = reinterpret_cast<const uint4*>(
          wlds + (((1 * 4 + k) * 8 + n) * 32 + lane) * 16);
      b1.u[0] = l1[0]; b1.u[1] = l1[1];
      b2.u[0] = l2[0]; b2.u[1] = l2[1];
      acc[n] = __builtin_amdgcn_wmma_f32_16x16x32_bf16(
          false, a1.v, false, b1.v, (short)0, acc[n], false, false);
      acc[n] = __builtin_amdgcn_wmma_f32_16x16x32_bf16(
          false, a2.v, false, b2.v, (short)0, acc[n], false, false);
    }
  }

  // Epilogue: C/D layout -> lane l, VGPR r: M = r + 8*half, N = n*16 + l%16.
#pragma unroll
  for (int n = 0; n < 8; ++n) {
    int col = n * 16 + (lane & 15);
#pragma unroll
    for (int r = 0; r < 8; ++r) {
      int orow = tile * 16 + r + half * 8;
      float x = acc[n][r];
      out[(long)orow * RGCN_D + col] = (x > 0.0f) ? x : x * RRELU_SLOPE;
    }
  }
}

// ---------------------------------------------------------------------------
extern "C" void kernel_launch(void* const* d_in, const int* in_sizes, int n_in,
                              void* d_out, int out_size, void* d_ws, size_t ws_size,
                              hipStream_t stream) {
  const float* ent  = (const float*)d_in[0];
  const float* rel  = (const float*)d_in[1];
  const float* norm = (const float*)d_in[2];
  const float* Wr0  = (const float*)d_in[3];
  const float* Wl0  = (const float*)d_in[4];
  const float* Wr1  = (const float*)d_in[5];
  const float* Wl1  = (const float*)d_in[6];
  const int*   src  = (const int*)d_in[7];
  const int*   dst  = (const int*)d_in[8];
  const int*   rid  = (const int*)d_in[9];

  int N = in_sizes[2];          // norm has N elements
  int E = in_sizes[7];          // src has E elements
  long ND = (long)N * RGCN_D;

  char* ws = (char*)d_ws;
  float*          h1     = (float*)ws;                         // N*D f32
  float*          agg    = (float*)(ws + ND * 4);              // N*D f32
  unsigned short* packed = (unsigned short*)(ws + 2 * ND * 4); // 64 KB

  float* outf = (float*)d_out;

  int ntiles = N / 16;                                         // 100000 -> 6250
  dim3 zgrid((unsigned)((ND + 255) / 256));
  dim3 sgrid((unsigned)(((long)E * 32 + 255) / 256));
  dim3 rgrid((32768 + 255) / 256);
  dim3 ggrid((unsigned)((ntiles + 7) / 8));

  // -------- layer 0 (h = ent_embed) --------
  rgcn_zero   <<<zgrid, 256, 0, stream>>>(agg, ND);
  rgcn_scatter<<<sgrid, 256, 0, stream>>>(ent, rel, src, dst, rid, agg, E);
  rgcn_repack <<<rgrid, 256, 0, stream>>>(Wr0, Wl0, packed);
  rgcn_gemm   <<<ggrid, 256, 0, stream>>>(agg, ent, norm, packed, h1, ntiles);

  // -------- layer 1 (h = h1) --------
  rgcn_zero   <<<zgrid, 256, 0, stream>>>(agg, ND);
  rgcn_scatter<<<sgrid, 256, 0, stream>>>(h1, rel, src, dst, rid, agg, E);
  rgcn_repack <<<rgrid, 256, 0, stream>>>(Wr1, Wl1, packed);
  rgcn_gemm   <<<ggrid, 256, 0, stream>>>(agg, h1, norm, packed, outf, ntiles);
}
